// HierarchicalSoftmax2_75969381531913
// MI455X (gfx1250) — matrix operand order repaired and checked by reading, not compiled, not run on hardware
//
#include <hip/hip_runtime.h>
#include <hip/hip_bf16.h>

// Problem constants (match reference)
#define B_TOK 4096
#define H_DIM 1024
#define C_CAT 64
#define N_SUB 786
#define N_PAD 800           // padded to 50 WMMA tiles of 16
#define KT    32            // K per v_wmma_f32_16x16x32_bf16
#define MT    32            // batch rows per block (2 WMMA m-tiles)
#define WAVES 10            // waves per block (wave32)
#define TPW   5             // N-tiles per wave; WAVES*TPW*16 == N_PAD

typedef __attribute__((ext_vector_type(4)))  unsigned int v4u;
typedef __attribute__((ext_vector_type(16))) __bf16       v16bf;
typedef __attribute__((ext_vector_type(8)))  float        v8f;

union Frag { v4u u[2]; v16bf bf; };

// fp32 -> bf16, round-to-nearest-even
__device__ __forceinline__ unsigned short f2bf(float x) {
    unsigned int u = __float_as_uint(x);
    u += 0x7FFFu + ((u >> 16) & 1u);
    return (unsigned short)(u >> 16);
}

// ---- Kernel 1: convert activations to bf16 -------------------------------
__global__ void cvt_inputs_kernel(const float* __restrict__ in,
                                  unsigned short* __restrict__ out) {
    int i = blockIdx.x * blockDim.x + threadIdx.x;   // grid sized exactly B*H
    out[i] = f2bf(in[i]);
}

// ---- Kernel 2: W_bottom [C,H,N] fp32 -> Wt [C,N_PAD,H] bf16 --------------
// LDS-tiled transpose: coalesced 128B reads (n fast), coalesced 64B writes (h fast).
__global__ __launch_bounds__(256)
void cvt_w_kernel(const float* __restrict__ w, unsigned short* __restrict__ wt) {
    __shared__ unsigned short tile[32][34];          // [n_local][h_local], padded
    const int c  = blockIdx.z;
    const int h0 = blockIdx.y * 32;
    const int n0 = blockIdx.x * 32;
    const int tx = threadIdx.x & 31;
    const int ty = threadIdx.x >> 5;                 // 0..7
#pragma unroll
    for (int p = 0; p < 32; p += 8) {
        int h = h0 + ty + p;
        int n = n0 + tx;
        float v = 0.f;                               // zero-pad n in [786,800)
        if (n < N_SUB) v = w[(c * H_DIM + h) * N_SUB + n];
        tile[tx][ty + p] = f2bf(v);
    }
    __syncthreads();
#pragma unroll
    for (int p = 0; p < 32; p += 8) {
        int n = n0 + ty + p;
        int h = h0 + tx;
        wt[(c * N_PAD + n) * H_DIM + h] = tile[ty + p][tx];
    }
}

// ---- Kernel 3: top-level gating softmax ----------------------------------
__global__ void top_gate_kernel(const float* __restrict__ x,
                                const float* __restrict__ wtop,
                                const float* __restrict__ btop,
                                float* __restrict__ gate) {
    int b = blockIdx.x;
    int c = threadIdx.x;                              // 64 threads
    const float* xr = x + b * H_DIM;
    float acc = btop[c];
    for (int h = 0; h < H_DIM; ++h) acc += xr[h] * wtop[h * C_CAT + c];
    __shared__ float sl[C_CAT];
    __shared__ float se[C_CAT];
    sl[c] = acc;
    __syncthreads();
    float mx = -3.4e38f;
    for (int j = 0; j < C_CAT; ++j) mx = fmaxf(mx, sl[j]);
    float e = __expf(acc - mx);
    se[c] = e;
    __syncthreads();
    float s = 0.f;
    for (int j = 0; j < C_CAT; ++j) s += se[j];
    gate[b * C_CAT + c] = e / s;
}

// ---- Kernel 4: fused bf16-WMMA GEMM + bias + softmax + gate scale --------
// Block = 32 batch rows x 800 (padded) columns for one category.
// Each wave: 2 m-tiles x 5 n-tiles => every B fragment feeds 2 WMMAs.
// Softmax is register-resident: shfl_xor butterfly within 16-lane halves,
// tiny LDS arrays to combine the 10 waves, stores straight from registers.
__global__ __launch_bounds__(WAVES * 32)
void moe_gemm_softmax_kernel(const unsigned short* __restrict__ Abf,  // [B,H] bf16
                             const unsigned short* __restrict__ Wt,   // [C,N_PAD,H] bf16
                             const float* __restrict__ bbot,          // [C,N]
                             const float* __restrict__ gate,          // [B,C]
                             float* __restrict__ out) {               // [B,C*N]
    __shared__ float s_pmax[WAVES][MT];
    __shared__ float s_psum[WAVES][MT];
    __shared__ float s_rowmax[MT];
    __shared__ float s_scale[MT];

    const int m0   = blockIdx.x * MT;
    const int c    = blockIdx.y;
    const int tid  = threadIdx.x;
    const int lane = tid & 31;
    const int wid  = tid >> 5;
    const int sel  = lane >> 4;            // half-wave: 0 or 1
    const int l16  = lane & 15;

    // ---------------- GEMM: D[32 x 800] = A[32 x 1024] * Wt^T -------------
    v8f acc[2][TPW] = {};

    // A fragment layout (16-bit A 16x32): lane<16 holds K {kk..+7, kk+16..+23},
    // lane>=16 holds K {kk+8..+15, kk+24..+31} for row m0 + l16 (+16 for mt=1).
    const unsigned short* arow0 = Abf + (m0 + l16) * H_DIM + sel * 8;
    const unsigned short* arow1 = arow0 + 16 * H_DIM;
    // B fragment: lane n holds column n, same K split as A per half-wave.
    const int n_base = wid * (TPW * 16) + l16;
    const unsigned short* brow0 = Wt + (c * N_PAD + n_base) * H_DIM + sel * 8;

    for (int kk = 0; kk < H_DIM; kk += KT) {
        Frag a0, a1;
        a0.u[0] = *reinterpret_cast<const v4u*>(arow0 + kk);
        a0.u[1] = *reinterpret_cast<const v4u*>(arow0 + kk + 16);
        a1.u[0] = *reinterpret_cast<const v4u*>(arow1 + kk);
        a1.u[1] = *reinterpret_cast<const v4u*>(arow1 + kk + 16);
#pragma unroll
        for (int t = 0; t < TPW; ++t) {
            const unsigned short* bp = brow0 + t * 16 * H_DIM + kk;
            Frag bfr;
            bfr.u[0] = *reinterpret_cast<const v4u*>(bp);
            bfr.u[1] = *reinterpret_cast<const v4u*>(bp + 16);
            acc[0][t] = __builtin_amdgcn_wmma_f32_16x16x32_bf16(
                false, a0.bf, false, bfr.bf, (short)0, acc[0][t], false, false);
            acc[1][t] = __builtin_amdgcn_wmma_f32_16x16x32_bf16(
                false, a1.bf, false, bfr.bf, (short)0, acc[1][t], false, false);
        }
    }

    // ---------------- bias add + pad mask (register-resident) -------------
    const float* bb = bbot + c * N_SUB;
    float bias[TPW];
    bool  valid[TPW];
#pragma unroll
    for (int t = 0; t < TPW; ++t) {
        int n = n_base + t * 16;
        valid[t] = (n < N_SUB);
        bias[t]  = valid[t] ? bb[n] : 0.0f;
    }
#pragma unroll
    for (int mt = 0; mt < 2; ++mt)
#pragma unroll
        for (int t = 0; t < TPW; ++t)
#pragma unroll
            for (int i = 0; i < 8; ++i)
                acc[mt][t][i] = valid[t] ? (acc[mt][t][i] + bias[t]) : -1e30f;

    // ---------------- per-row max: shfl butterfly + cross-wave LDS --------
#pragma unroll
    for (int mt = 0; mt < 2; ++mt) {
#pragma unroll
        for (int i = 0; i < 8; ++i) {
            float m = acc[mt][0][i];
#pragma unroll
            for (int t = 1; t < TPW; ++t) m = fmaxf(m, acc[mt][t][i]);
#pragma unroll
            for (int off = 1; off < 16; off <<= 1)
                m = fmaxf(m, __shfl_xor(m, off, 32));
            if (l16 == 0) s_pmax[wid][mt * 16 + sel * 8 + i] = m;
        }
    }
    __syncthreads();
    if (tid < MT) {
        float m2 = -3.4e38f;
        for (int w = 0; w < WAVES; ++w) m2 = fmaxf(m2, s_pmax[w][tid]);
        s_rowmax[tid] = m2;
    }
    __syncthreads();

    // ---------------- exp + per-row sum -----------------------------------
#pragma unroll
    for (int mt = 0; mt < 2; ++mt) {
#pragma unroll
        for (int i = 0; i < 8; ++i) {
            float rm = s_rowmax[mt * 16 + sel * 8 + i];
            float s = 0.f;
#pragma unroll
            for (int t = 0; t < TPW; ++t) {
                float e = __expf(acc[mt][t][i] - rm);
                acc[mt][t][i] = e;
                s += e;
            }
#pragma unroll
            for (int off = 1; off < 16; off <<= 1)
                s += __shfl_xor(s, off, 32);
            if (l16 == 0) s_psum[wid][mt * 16 + sel * 8 + i] = s;
        }
    }
    __syncthreads();
    if (tid < MT) {
        float s2 = 0.f;
        for (int w = 0; w < WAVES; ++w) s2 += s_psum[w][tid];
        s_scale[tid] = gate[(m0 + tid) * C_CAT + c] / s2;
    }
    __syncthreads();

    // ---------------- store from registers (64B contiguous per half) ------
    float* obase = out + (size_t)c * N_SUB;
#pragma unroll
    for (int mt = 0; mt < 2; ++mt) {
#pragma unroll
        for (int t = 0; t < TPW; ++t) {
            int n = n_base + t * 16;
            if (n < N_SUB) {
#pragma unroll
                for (int i = 0; i < 8; ++i) {
                    int row = mt * 16 + sel * 8 + i;
                    obase[(size_t)(m0 + row) * (C_CAT * N_SUB) + n] =
                        acc[mt][t][i] * s_scale[row];
                }
            }
        }
    }
}

// ---------------------------------------------------------------------------
extern "C" void kernel_launch(void* const* d_in, const int* in_sizes, int n_in,
                              void* d_out, int out_size, void* d_ws, size_t ws_size,
                              hipStream_t stream) {
    (void)in_sizes; (void)n_in; (void)out_size; (void)ws_size;

    const float* x    = (const float*)d_in[0];   // [B,H]
    const float* wtop = (const float*)d_in[1];   // [H,C]
    const float* btop = (const float*)d_in[2];   // [C]
    const float* wbot = (const float*)d_in[3];   // [C,H,N]
    const float* bbot = (const float*)d_in[4];   // [C,N]
    float* out = (float*)d_out;                  // [B, C*N]

    // Workspace layout (~114 MB): A_bf16 | Wt_bf16 | gate
    char* ws = (char*)d_ws;
    unsigned short* Abf = (unsigned short*)ws;
    unsigned short* Wt  = (unsigned short*)(ws + (size_t)B_TOK * H_DIM * 2);
    float* gatep        = (float*)(ws + (size_t)B_TOK * H_DIM * 2
                                      + (size_t)C_CAT * N_PAD * H_DIM * 2);

    cvt_inputs_kernel<<<(B_TOK * H_DIM) / 256, 256, 0, stream>>>(x, Abf);
    dim3 wgrid(N_PAD / 32, H_DIM / 32, C_CAT);
    cvt_w_kernel<<<wgrid, 256, 0, stream>>>(wbot, Wt);
    top_gate_kernel<<<B_TOK, C_CAT, 0, stream>>>(x, wtop, btop, gatep);

    dim3 grid(B_TOK / MT, C_CAT);
    moe_gemm_softmax_kernel<<<grid, WAVES * 32, 0, stream>>>(Abf, Wt, bbot, gatep, out);
}
